// SimpleMoELayer_28217935134730
// MI455X (gfx1250) — matrix operand order repaired and checked by reading, not compiled, not run on hardware
//
#include <hip/hip_runtime.h>
#include <hip/hip_bf16.h>
#include <stdint.h>

// ---------------- Problem constants (from reference) ----------------
#define HIDDEN 2048
#define FFN    8192
#define NEXP   8
#define TOPK   2
#define TOKENS 4096                 // 2 * 2048
#define NSLOTS (TOKENS * TOPK)      // 8192 routed (token,expert) assignments

// ---------------- GEMM tiling ----------------
#define BM   128
#define BN   128
#define BK   32
#define KPAD 48                     // LDS row stride in ushorts = 96B (16B aligned rows)

typedef __attribute__((ext_vector_type(16))) __bf16 v16bf;
typedef __attribute__((ext_vector_type(2)))  __bf16 v2bf;
typedef __attribute__((ext_vector_type(8)))  float  v8f;

union Frag { uint4 q[2]; v16bf v; };   // 32 bytes: one WMMA A/B operand

// ---------------- TDM descriptor types (probe-verified builtin signatures) ----
typedef unsigned int u32t;
typedef unsigned long long u64t;
typedef __attribute__((ext_vector_type(4))) u32t tdm_v4u;
typedef __attribute__((ext_vector_type(4))) int  tdm_v4i;
typedef __attribute__((ext_vector_type(8))) int  tdm_v8i;

#if defined(__clang_major__) && (__clang_major__ >= 21)
#define USE_TDM 1
#else
#define USE_TDM 0
#endif

// Issue a 2D TDM tile load: rows x cols elements of 2B data, row stride = srcStride
// elements, destination LDS rows padded 64B->96B (pad_interval=3 -> 16 DWORDs,
// pad_amount=7 -> 8 DWORDs) to match the KPAD=48-ushort LDS layout.
__device__ __forceinline__ void tdm_load_tile_bf16(
    unsigned lds_addr, u64t gaddr, u32t rows_avail, u32t src_stride_elems) {
#if USE_TDM
  tdm_v4u g0;
  g0[0] = 1u;                                        // count=1 valid descriptor
  g0[1] = lds_addr;                                  // LDS byte address
  g0[2] = (u32t)gaddr;                               // global_addr[31:0]
  g0[3] = (u32t)((gaddr >> 32) & 0x1FFFFFFull) | (2u << 30);  // addr[56:32] | type=2

  tdm_v8i g1;
  u32t tdim0 = FFN;                                  // tensor dim0 (elems, OOB bound)
  u32t tdim1 = rows_avail;                           // remaining valid rows -> OOB rows read 0
  g1[0] = (int)((1u << 16)                           // data_size = 1 (2 bytes)
              | (1u << 20)                           // pad_enable
              | (3u << 22)                           // pad_interval: 16 DWORDs (=64B row)
              | (7u << 25));                         // pad_amount: 8 DWORDs (=32B pad)
  g1[1] = (int)((tdim0 & 0xFFFFu) << 16);            // tensor_dim0[15:0] at bits[63:48]
  g1[2] = (int)((tdim0 >> 16) | ((tdim1 & 0xFFFFu) << 16));
  g1[3] = (int)((tdim1 >> 16) | ((u32t)BK << 16));   // tile_dim0 = BK
  g1[4] = (int)((u32t)BM);                           // tile_dim1 = BM, tile_dim2 = 0
  g1[5] = (int)src_stride_elems;                     // tensor_dim0_stride[31:0]
  g1[6] = 0;                                         // stride hi + dim1_stride lo
  g1[7] = 0;

  tdm_v4i z4 = {0, 0, 0, 0};
#if (__clang_major__ >= 23)
  tdm_v8i z8 = {0, 0, 0, 0, 0, 0, 0, 0};
  __builtin_amdgcn_tensor_load_to_lds(g0, g1, z4, z4, z8, 0);
#else
  __builtin_amdgcn_tensor_load_to_lds(g0, g1, z4, z4, 0);
#endif
#endif
}

// ---------------- helpers ----------------
// Native fptrunc f32->bf16: lowers to the CDNA5 hardware converter
// (v_cvt_pk_bf16_f32 class) instead of manual integer RNE (saves ~10 VALU/pair).
union BfBits { __bf16 h; unsigned short u; };

__device__ __forceinline__ unsigned short f32_to_bf16(float f) {
  BfBits b; b.h = (__bf16)f;
  return b.u;
}

__device__ __forceinline__ unsigned pack2_bf16(float lo, float hi) {
  union { v2bf v; unsigned u; } p;
  p.v = (v2bf){(__bf16)lo, (__bf16)hi};
  return p.u;
}

__device__ __forceinline__ uint2 pack4_bf16(float4 f) {
  uint2 r;
  r.x = pack2_bf16(f.x, f.y);
  r.y = pack2_bf16(f.z, f.w);
  return r;
}

__device__ __forceinline__ float gelu_tanh(float v) {   // jax.nn.gelu (approximate)
  float c = 0.7978845608028654f * (v + 0.044715f * v * v * v);
  return 0.5f * v * (1.0f + tanhf(c));
}

// meta layout (ints at ws offset 0): [0..7]=counts  [8..15]=fill  [16..23]=offsets

// ---------------- kernel 0: zero output + counters ----------------
__global__ void moe_init_kernel(float* __restrict__ out, int* __restrict__ meta) {
  size_t i = (size_t)blockIdx.x * blockDim.x + threadIdx.x;
  size_t n4 = (size_t)TOKENS * HIDDEN / 4;
  if (i < n4) ((float4*)out)[i] = make_float4(0.f, 0.f, 0.f, 0.f);
  if (blockIdx.x == 0 && threadIdx.x < 32) meta[threadIdx.x] = 0;
}

// ---------------- kernel 1: router logits + top-2 softmax ----------------
__global__ __launch_bounds__(256) void moe_router_kernel(
    const float* __restrict__ x, const float* __restrict__ rw,
    int* __restrict__ top_i, float* __restrict__ top_g, int* __restrict__ meta) {
  __shared__ float sred[256];
  __shared__ float logits[NEXP];
  int t = blockIdx.x;
  const float* xr = x + (size_t)t * HIDDEN;
  float acc[NEXP];
#pragma unroll
  for (int e = 0; e < NEXP; ++e) acc[e] = 0.f;
  for (int h = threadIdx.x; h < HIDDEN; h += 256) {
    float xv = xr[h];
    const float* r = rw + (size_t)h * NEXP;
#pragma unroll
    for (int e = 0; e < NEXP; ++e) acc[e] = __builtin_fmaf(xv, r[e], acc[e]);
  }
  for (int e = 0; e < NEXP; ++e) {
    sred[threadIdx.x] = acc[e];
    __syncthreads();
    for (int s = 128; s > 0; s >>= 1) {
      if (threadIdx.x < s) sred[threadIdx.x] += sred[threadIdx.x + s];
      __syncthreads();
    }
    if (threadIdx.x == 0) logits[e] = sred[0];
    __syncthreads();
  }
  if (threadIdx.x == 0) {
    int bi = 0; float bv = logits[0];
    for (int e = 1; e < NEXP; ++e) if (logits[e] > bv) { bv = logits[e]; bi = e; }
    int si = (bi == 0) ? 1 : 0; float sv = logits[si];
    for (int e = 0; e < NEXP; ++e) if (e != bi && logits[e] > sv) { sv = logits[e]; si = e; }
    float e1  = __expf(sv - bv);
    float den = 1.0f + e1;
    top_i[t * 2 + 0] = bi; top_g[t * 2 + 0] = 1.0f / den;
    top_i[t * 2 + 1] = si; top_g[t * 2 + 1] = e1 / den;
    atomicAdd(&meta[bi], 1);
    atomicAdd(&meta[si], 1);
  }
}

// ---------------- kernel 2: exclusive prefix-sum over 8 expert counts ----------------
__global__ void moe_offsets_kernel(int* __restrict__ meta) {
  if (threadIdx.x == 0) {
    int run = 0;
    for (int e = 0; e < NEXP; ++e) { meta[16 + e] = run; run += meta[e]; meta[8 + e] = 0; }
  }
}

// ---------------- kernel 3: build compact per-expert slot lists ----------------
__global__ void moe_scatter_kernel(
    const int* __restrict__ top_i, const float* __restrict__ top_g,
    int* __restrict__ meta, int* __restrict__ slot_token, float* __restrict__ slot_gate) {
  int t = blockIdx.x * blockDim.x + threadIdx.x;
  if (t >= TOKENS) return;
#pragma unroll
  for (int k = 0; k < TOPK; ++k) {
    int e    = top_i[t * 2 + k];
    int pos  = atomicAdd(&meta[8 + e], 1);
    int slot = meta[16 + e] + pos;
    slot_token[slot] = t;
    slot_gate[slot]  = top_g[t * 2 + k];
  }
}

// ---------------- kernel 4: h = gelu(Xg @ w1[e])  (routed, bf16 WMMA) ----------------
__global__ __launch_bounds__(256) void moe_gemm1_kernel(
    const float* __restrict__ x, const float* __restrict__ w1,
    const int* __restrict__ meta, const int* __restrict__ slot_token,
    unsigned short* __restrict__ hbuf) {
  int e     = blockIdx.z;
  int count = meta[e];
  if (count == 0) return;
  int base   = meta[16 + e];
  int tile_t = blockIdx.y;
  if (tile_t * BM >= count) return;
  int nbase = blockIdx.x * BN;
  const float* W = w1 + (size_t)e * HIDDEN * FFN;

  __shared__ __align__(16) unsigned short As[BM][KPAD];   // [m][k]  bf16
  __shared__ __align__(16) unsigned short Bs[BN][KPAD];   // [n][k]  bf16 (transposed)

  int tid  = threadIdx.x;
  int lane = tid & 31, wv = tid >> 5;
  int wm   = wv & 3,   wn = wv >> 2;           // 4x2 wave grid -> 128x128 block tile

  // A-loader: 128 rows x 8 float4 = 1024 quads, 4 per thread; gather token ids once
  int rowA[4], colqA[4], tokA[4];
#pragma unroll
  for (int i = 0; i < 4; ++i) {
    int lin  = tid + i * 256;
    rowA[i]  = lin >> 3;
    colqA[i] = lin & 7;
    int g = tile_t * BM + rowA[i];
    if (g >= count) g = count - 1;             // clamped gather, stores are guarded
    tokA[i] = slot_token[base + g];
  }

  v8f c[2][4];
#pragma unroll
  for (int mi = 0; mi < 2; ++mi)
#pragma unroll
    for (int ni = 0; ni < 4; ++ni) c[mi][ni] = (v8f){0.f,0.f,0.f,0.f,0.f,0.f,0.f,0.f};

  int kg = lane >> 4;        // K interleave group (ISA A-layout: lanes 16-31 hold K+8/K+24)
  int ml = lane & 15;

  for (int kt = 0; kt < HIDDEN / BK; ++kt) {
    int k0 = kt * BK;
    // stage A: gathered x rows, fp32 -> bf16 in registers (hw cvt)
#pragma unroll
    for (int i = 0; i < 4; ++i) {
      float4 f = *(const float4*)(x + (size_t)tokA[i] * HIDDEN + k0 + colqA[i] * 4);
      *(uint2*)&As[rowA[i]][colqA[i] * 4] = pack4_bf16(f);
    }
    // stage B: w1 tile [BKxBN], fp32 -> bf16, transposed into Bs[n][k]
#pragma unroll
    for (int i = 0; i < 4; ++i) {
      int lin = tid + i * 256;
      int kr  = lin >> 5;
      int nc  = (lin & 31) * 4;
      float4 f = *(const float4*)(W + (size_t)(k0 + kr) * FFN + nbase + nc);
      Bs[nc + 0][kr] = f32_to_bf16(f.x);
      Bs[nc + 1][kr] = f32_to_bf16(f.y);
      Bs[nc + 2][kr] = f32_to_bf16(f.z);
      Bs[nc + 3][kr] = f32_to_bf16(f.w);
    }
    __syncthreads();
    if (kt + 1 < HIDDEN / BK) {   // L2 prefetch of next K-tile -> global_prefetch_b8
      __builtin_prefetch(x + (size_t)tokA[0] * HIDDEN + k0 + BK, 0, 1);
      __builtin_prefetch(W + (size_t)(k0 + BK) * FFN + nbase + (lane * 4), 0, 1);
    }
    Frag a[2], b[4];
#pragma unroll
    for (int mi = 0; mi < 2; ++mi) {
      int r = wm * 32 + mi * 16 + ml;
      a[mi].q[0] = *(const uint4*)&As[r][kg * 8];        // K = kg*8 .. +7
      a[mi].q[1] = *(const uint4*)&As[r][16 + kg * 8];   // K = 16+kg*8 .. +7
    }
#pragma unroll
    for (int ni = 0; ni < 4; ++ni) {
      int n = wn * 64 + ni * 16 + ml;
      b[ni].q[0] = *(const uint4*)&Bs[n][kg * 8];        // K = kg*8 .. +7
      b[ni].q[1] = *(const uint4*)&Bs[n][16 + kg * 8];   // K = 16+kg*8 .. +7
    }
#pragma unroll
    for (int mi = 0; mi < 2; ++mi)
#pragma unroll
      for (int ni = 0; ni < 4; ++ni)
        c[mi][ni] = __builtin_amdgcn_wmma_f32_16x16x32_bf16(
            false, a[mi].v, false, b[ni].v, (short)0, c[mi][ni], false, false);
    __syncthreads();
  }

  // epilogue: GELU, pack bf16, store routed h
  int mh = lane >> 4;
#pragma unroll
  for (int mi = 0; mi < 2; ++mi) {
#pragma unroll
    for (int r = 0; r < 8; ++r) {
      int lrow = wm * 32 + mi * 16 + mh * 8 + r;
      int g    = tile_t * BM + lrow;
      if (g < count) {
#pragma unroll
        for (int ni = 0; ni < 4; ++ni) {
          int col = nbase + wn * 64 + ni * 16 + ml;
          hbuf[(size_t)(base + g) * FFN + col] = f32_to_bf16(gelu_tanh(c[mi][ni][r]));
        }
      }
    }
  }
}

// ---------------- kernel 5: out += gate * (h @ w2[e])  (routed, bf16 WMMA, TDM A) ----
__global__ __launch_bounds__(256) void moe_gemm2_kernel(
    const unsigned short* __restrict__ hbuf, const float* __restrict__ w2,
    const int* __restrict__ meta, const int* __restrict__ slot_token,
    const float* __restrict__ slot_gate, float* __restrict__ out) {
  int e     = blockIdx.z;
  int count = meta[e];
  if (count == 0) return;
  int base   = meta[16 + e];
  int tile_t = blockIdx.y;
  if (tile_t * BM >= count) return;
  int nbase = blockIdx.x * BN;
  const float* W = w2 + (size_t)e * FFN * HIDDEN;

  __shared__ __align__(16) unsigned short As[BM][KPAD];
  __shared__ __align__(16) unsigned short Bs[BN][KPAD];

  int tid  = threadIdx.x;
  int lane = tid & 31, wv = tid >> 5;
  int wm   = wv & 3,   wn = wv >> 2;

  // TDM source: contiguous slot rows of hbuf (2D tile BM x BK, row stride FFN).
  unsigned    as_lds     = (unsigned)(uintptr_t)(void*)&As[0][0]; // low 32b = LDS offset
  u64t        tile_gaddr = (u64t)(uintptr_t)(hbuf + (size_t)(base + tile_t * BM) * FFN);
  u32t        rows_avail = (u32t)(count - tile_t * BM);  // OOB rows read as zero by TDM

  v8f c[2][4];
#pragma unroll
  for (int mi = 0; mi < 2; ++mi)
#pragma unroll
    for (int ni = 0; ni < 4; ++ni) c[mi][ni] = (v8f){0.f,0.f,0.f,0.f,0.f,0.f,0.f,0.f};

  int kg = lane >> 4, ml = lane & 15;

#if !USE_TDM
  // fallback loader state (only compiled on unknown toolchains)
  int rowA[2], colqA[2]; size_t srcA[2];
#pragma unroll
  for (int i = 0; i < 2; ++i) {
    int lin  = tid + i * 256;
    rowA[i]  = lin >> 2;
    colqA[i] = (lin & 3) * 8;
    int g = tile_t * BM + rowA[i];
    if (g >= count) g = count - 1;
    srcA[i] = (size_t)(base + g) * FFN + colqA[i];
  }
#endif

  for (int kt = 0; kt < FFN / BK; ++kt) {
    int k0 = kt * BK;
#if USE_TDM
    if (wv == 0) {  // one wave issues the DMA; TDM ignores EXEC, done once per block
      tdm_load_tile_bf16(as_lds, tile_gaddr + (u64t)k0 * 2u, rows_avail, FFN);
    }
#else
#pragma unroll
    for (int i = 0; i < 2; ++i)
      *(uint4*)&As[rowA[i]][colqA[i]] = *(const uint4*)&hbuf[srcA[i] + k0];
#endif
    // stage B: w2 tile [BKxBN], fp32 -> bf16, transposed into Bs[n][k]
#pragma unroll
    for (int i = 0; i < 4; ++i) {
      int lin = tid + i * 256;
      int kr  = lin >> 5;
      int nc  = (lin & 31) * 4;
      float4 f = *(const float4*)(W + (size_t)(k0 + kr) * HIDDEN + nbase + nc);
      Bs[nc + 0][kr] = f32_to_bf16(f.x);
      Bs[nc + 1][kr] = f32_to_bf16(f.y);
      Bs[nc + 2][kr] = f32_to_bf16(f.z);
      Bs[nc + 3][kr] = f32_to_bf16(f.w);
    }
#if USE_TDM
    if (wv == 0) __builtin_amdgcn_s_wait_tensorcnt(0);  // A tile resident in LDS
#endif
    __syncthreads();
    if (kt + 1 < FFN / BK) {
      __builtin_prefetch(W + (size_t)(k0 + BK) * HIDDEN + nbase + (lane * 4), 0, 1);
    }
    Frag a[2], b[4];
#pragma unroll
    for (int mi = 0; mi < 2; ++mi) {
      int r = wm * 32 + mi * 16 + ml;
      a[mi].q[0] = *(const uint4*)&As[r][kg * 8];
      a[mi].q[1] = *(const uint4*)&As[r][16 + kg * 8];
    }
#pragma unroll
    for (int ni = 0; ni < 4; ++ni) {
      int n = wn * 64 + ni * 16 + ml;
      b[ni].q[0] = *(const uint4*)&Bs[n][kg * 8];
      b[ni].q[1] = *(const uint4*)&Bs[n][16 + kg * 8];
    }
#pragma unroll
    for (int mi = 0; mi < 2; ++mi)
#pragma unroll
      for (int ni = 0; ni < 4; ++ni)
        c[mi][ni] = __builtin_amdgcn_wmma_f32_16x16x32_bf16(
            false, a[mi].v, false, b[ni].v, (short)0, c[mi][ni], false, false);
    __syncthreads();
  }

  // epilogue: gate-scale and atomically combine both experts per token
  int mh = lane >> 4;
#pragma unroll
  for (int mi = 0; mi < 2; ++mi) {
#pragma unroll
    for (int r = 0; r < 8; ++r) {
      int lrow = wm * 32 + mi * 16 + mh * 8 + r;
      int g    = tile_t * BM + lrow;
      if (g < count) {
        int   slot = base + g;
        int   t    = slot_token[slot];
        float gt   = slot_gate[slot];
#pragma unroll
        for (int ni = 0; ni < 4; ++ni) {
          int col = nbase + wn * 64 + ni * 16 + ml;
          atomicAdd(&out[(size_t)t * HIDDEN + col], gt * c[mi][ni][r]);
        }
      }
    }
  }
}

// ---------------- launch ----------------
extern "C" void kernel_launch(void* const* d_in, const int* in_sizes, int n_in,
                              void* d_out, int out_size, void* d_ws, size_t ws_size,
                              hipStream_t stream) {
  const float* x  = (const float*)d_in[0];   // [2,2048,2048]
  const float* rw = (const float*)d_in[1];   // [2048,8]
  const float* w1 = (const float*)d_in[2];   // [8,2048,8192]
  const float* w2 = (const float*)d_in[3];   // [8,8192,2048]
  float* out = (float*)d_out;                // [2,2048,2048] fp32

  // workspace layout (needs ~134.3 MB):
  //   [0,128)                        meta ints: counts/fill/offsets
  //   [128, +32KB)                   top_i  [T][2] int
  //   next 32KB                      top_g  [T][2] float
  //   next 32KB                      slot_token [2T] int
  //   next 32KB                      slot_gate  [2T] float
  //   aligned 256                    hbuf [2T][FFN] bf16 (128 MiB)
  char* ws = (char*)d_ws;
  int*            meta       = (int*)ws;
  int*            top_i      = (int*)(ws + 128);
  float*          top_g      = (float*)(ws + 128 + (size_t)TOKENS * 2 * 4);
  int*            slot_token = (int*)(ws + 128 + (size_t)TOKENS * 4 * 4);
  float*          slot_gate  = (float*)(ws + 128 + (size_t)TOKENS * 6 * 4);
  unsigned short* hbuf       = (unsigned short*)(ws + ((128 + (size_t)TOKENS * 8 * 4 + 255) & ~(size_t)255));

  {
    size_t n4 = (size_t)TOKENS * HIDDEN / 4;
    moe_init_kernel<<<(unsigned)((n4 + 255) / 256), 256, 0, stream>>>(out, meta);
  }
  moe_router_kernel<<<TOKENS, 256, 0, stream>>>(x, rw, top_i, top_g, meta);
  moe_offsets_kernel<<<1, 32, 0, stream>>>(meta);
  moe_scatter_kernel<<<TOKENS / 256, 256, 0, stream>>>(top_i, top_g, meta, slot_token, slot_gate);

  dim3 g1(FFN / BN, TOKENS / BM, NEXP);     // (64, 32, 8); tiles past count exit early
  moe_gemm1_kernel<<<g1, 256, 0, stream>>>(x, w1, meta, slot_token, hbuf);

  dim3 g2(HIDDEN / BN, TOKENS / BM, NEXP);  // (16, 32, 8)
  moe_gemm2_kernel<<<g2, 256, 0, stream>>>(hbuf, w2, meta, slot_token, slot_gate, out);
}
